// _GATv2Stack_48455821033915
// MI455X (gfx1250) — matrix operand order repaired and checked by reading, not compiled, not run on hardware
//
#include <hip/hip_runtime.h>
#include <hip/hip_bf16.h>
#include <math.h>

// Problem constants (match reference)
#define NNODES 50000
#define NEDGES 800000
#define E2TOT  (NNODES + NEDGES)   // edges + self loops
#define HIDDIM 128
#define HDC    32
#define NHEADS 4
#define EDIM   16

typedef __bf16 bf16_t;
typedef __attribute__((ext_vector_type(16))) __bf16 v16bf;
typedef __attribute__((ext_vector_type(8)))  float  v8f;

// ---------------- small helpers ----------------
__device__ __forceinline__ unsigned f2ord(float f) {
  unsigned u = __float_as_uint(f);
  return (u & 0x80000000u) ? ~u : (u | 0x80000000u);
}
__device__ __forceinline__ float ord2f(unsigned u) {
  unsigned v = (u & 0x80000000u) ? (u & 0x7fffffffu) : ~u;
  return __uint_as_float(v);
}
#define ORD_NEG_INF 0x007FFFFFu   // f2ord(-inf)

__global__ void k_fill_f32(float* p, int n, float v) {
  int i = blockIdx.x * blockDim.x + threadIdx.x;
  if (i < n) p[i] = v;
}
__global__ void k_fill_u32(unsigned* p, int n, unsigned v) {
  int i = blockIdx.x * blockDim.x + threadIdx.x;
  if (i < n) p[i] = v;
}
__global__ void k_f2bf(const float* __restrict__ s, bf16_t* __restrict__ d, int n) {
  int i = blockIdx.x * blockDim.x + threadIdx.x;
  if (i < n) d[i] = (bf16_t)s[i];
}
// Wt[n][k] = (bf16) W[k][n]   (W is K x Ncols row-major)
__global__ void k_w_to_bf_t(const float* __restrict__ W, bf16_t* __restrict__ Wt,
                            int K, int Ncols) {
  int i = blockIdx.x * blockDim.x + threadIdx.x;
  if (i >= K * Ncols) return;
  int n = i / K, k = i % K;
  Wt[i] = (bf16_t)W[k * Ncols + n];
}

// ------- bf16 WMMA GEMM: C[M,Nc] = A[M,K] @ Wt[Nc,K]^T + bias -------------
// One wave computes a 16 x (16*NTILES) output strip; the A fragment is loaded
// once per k-step and reused by NTILES back-to-back v_wmma_f32_16x16x32_bf16.
// NTILES is a compile-time constant -> fully unrolled, no branches, fixed
// accumulator registers, clause-batched loads.
// A row-major [M,K] bf16; Wt row-major [Nc,K] bf16 (i.e. W transposed).
template <int NTILES>
__global__ __launch_bounds__(32)
void k_gemm_bf16(const bf16_t* __restrict__ A, const bf16_t* __restrict__ Wt,
                 const float* __restrict__ bias, float* __restrict__ C,
                 int M, int K, int Nc) {
  const int lane = threadIdx.x;     // 0..31 (wave32)
  const int half = lane >> 4;       // 0 | 1
  const int l16  = lane & 15;
  const int mt  = blockIdx.x;
  const int nt0 = blockIdx.y * NTILES;
  const bf16_t* arow = A + (size_t)(mt * 16 + l16) * K;
  const bf16_t* brow[NTILES];
#pragma unroll
  for (int j = 0; j < NTILES; ++j)
    brow[j] = Wt + (size_t)((nt0 + j) * 16 + l16) * K;
  v8f acc[NTILES];
#pragma unroll
  for (int j = 0; j < NTILES; ++j) acc[j] = (v8f){};
  for (int kb = 0; kb < K; kb += 32) {
    union { unsigned u[8]; v16bf v; } a;
#pragma unroll
    for (int r = 0; r < 8; ++r) {
      // A 16x32 bf16 layout: lane<16 -> K base 0, lane>=16 -> +8; VGPR r<4 -> +0, r>=4 -> +16
      a.u[r] = *(const unsigned*)(arow + kb + ((r & 4) << 2) + half * 8 + ((r & 3) << 1));
    }
#pragma unroll
    for (int j = 0; j < NTILES; ++j) {
      union { unsigned u[8]; v16bf v; } b;
#pragma unroll
      for (int r = 0; r < 8; ++r) {
        // B 32x16 bf16 layout: lane<16 -> K base 0, lane>=16 -> +8; VGPR r<4 -> +0, r>=4 -> +16
        b.u[r] = *(const unsigned*)(brow[j] + kb + ((r & 4) << 2) + half * 8 + ((r & 3) << 1));
      }
      acc[j] = __builtin_amdgcn_wmma_f32_16x16x32_bf16(false, a.v, false, b.v,
                                                       (short)0, acc[j], false, false);
    }
  }
#pragma unroll
  for (int j = 0; j < NTILES; ++j) {
    int ncol = (nt0 + j) * 16 + l16;
    float bv = bias ? bias[ncol] : 0.0f;
#pragma unroll
    for (int r = 0; r < 8; ++r) {
      int mrow = mt * 16 + r + half * 8;  // C layout: VGPR r holds M=r (+8 for lanes 16..31)
      C[(size_t)mrow * Nc + ncol] = acc[j][r] + bv;
    }
  }
}

// ---------------- edge preprocessing (self-loop attr = mean of incoming) ---
__global__ void k_deg(const int* __restrict__ dst, float* __restrict__ deg, int E) {
  int i = blockIdx.x * blockDim.x + threadIdx.x;
  if (i < E) atomicAdd(&deg[dst[i]], 1.0f);
}
__global__ void k_ea_acc(const int* __restrict__ dst, const float* __restrict__ ea,
                         float* __restrict__ loop_ea, int E) {
  int i = blockIdx.x * blockDim.x + threadIdx.x;
  if (i >= E * EDIM) return;
  int e = i >> 4, d = i & 15;
  atomicAdd(&loop_ea[(size_t)dst[e] * EDIM + d], ea[i]);
}
__global__ void k_loop_div(float* __restrict__ loop_ea, const float* __restrict__ deg, int n) {
  int i = blockIdx.x * blockDim.x + threadIdx.x;
  if (i >= n) return;
  loop_ea[i] /= fmaxf(deg[i >> 4], 1.0f);
}

// ---------------- edge pass 1: GATv2 logits + segment max -----------------
// one wave per edge; lane = channel within a head; loop over 4 heads
__global__ __launch_bounds__(256)
void k_edge_logits(const int* __restrict__ ei, const float* __restrict__ ea2,
                   const float* __restrict__ We, const float* __restrict__ xl,
                   const float* __restrict__ xr, const float* __restrict__ att,
                   float* __restrict__ logits, unsigned* __restrict__ maxkey,
                   int E, int E2) {
  int wid = blockIdx.x * (blockDim.x >> 5) + (threadIdx.x >> 5);
  int lane = threadIdx.x & 31;
  if (wid >= E2) return;
  int s, d;
  if (wid < E) { s = ei[wid]; d = ei[E + wid]; } else { s = d = wid - E; }
  float eav = (lane < EDIM) ? ea2[(size_t)wid * EDIM + lane] : 0.0f;
#pragma unroll
  for (int h = 0; h < NHEADS; ++h) {
    int c = h * HDC + lane;
    float ep = 0.0f;
#pragma unroll
    for (int k = 0; k < EDIM; ++k)
      ep = fmaf(__shfl(eav, k, 32), We[k * HIDDIM + c], ep);
    float m = xl[(size_t)s * HIDDIM + c] + xr[(size_t)d * HIDDIM + c] + ep;
    m = (m > 0.0f) ? m : 0.2f * m;                 // LeakyReLU(0.2)
    float v = m * att[c];
#pragma unroll
    for (int off = 16; off > 0; off >>= 1) v += __shfl_down(v, off, 32);
    if (lane == 0) {
      logits[(size_t)wid * NHEADS + h] = v;
      atomicMax(&maxkey[(size_t)d * NHEADS + h], f2ord(v));
    }
  }
}

// ---------------- edge pass 2: exp + segment sum --------------------------
__global__ void k_edge_exp(const int* __restrict__ ei, float* __restrict__ logits,
                           const unsigned* __restrict__ maxkey, float* __restrict__ den,
                           int E, int E2) {
  int i = blockIdx.x * blockDim.x + threadIdx.x;
  if (i >= E2 * NHEADS) return;
  int e = i >> 2, h = i & 3;
  int d = (e < E) ? ei[E + e] : e - E;
  float ex = __expf(logits[i] - ord2f(maxkey[d * NHEADS + h]));
  logits[i] = ex;
  atomicAdd(&den[d * NHEADS + h], ex);
}

// ---------------- edge pass 3: out[dst] += alpha * xl[src] ----------------
__global__ __launch_bounds__(256)
void k_edge_aggr(const int* __restrict__ ei, const float* __restrict__ logits,
                 const float* __restrict__ den, const float* __restrict__ xl,
                 float* __restrict__ t, int E, int E2, int concat) {
  int wid = blockIdx.x * (blockDim.x >> 5) + (threadIdx.x >> 5);
  int lane = threadIdx.x & 31;
  if (wid >= E2) return;
  int s, d;
  if (wid < E) { s = ei[wid]; d = ei[E + wid]; } else { s = d = wid - E; }
#pragma unroll
  for (int h = 0; h < NHEADS; ++h) {
    float alpha = logits[(size_t)wid * NHEADS + h] / den[d * NHEADS + h];
    float val = alpha * xl[(size_t)s * HIDDIM + h * HDC + lane];
    if (concat) atomicAdd(&t[(size_t)d * HIDDIM + h * HDC + lane], val);
    else        atomicAdd(&t[(size_t)d * HDC + lane], 0.25f * val);  // head mean
  }
}

// t prefilled with bias (aggregation adds on top)
__global__ void k_prefill_bias(float* __restrict__ t, const float* __restrict__ bias,
                               int n, int C) {
  int i = blockIdx.x * blockDim.x + threadIdx.x;
  if (i < n) t[i] = bias[i % C];
}

// ---------------- BatchNorm (training stats) + GELU + residual ------------
__global__ __launch_bounds__(256)
void k_bnstat(const float* __restrict__ t, float* __restrict__ sums, int Nr, int C) {
  int c = threadIdx.x % C;
  int rpb = blockDim.x / C;
  int r = blockIdx.x * rpb + threadIdx.x / C;
  int stride = gridDim.x * rpb;
  float s = 0.0f, s2 = 0.0f;
  for (; r < Nr; r += stride) {
    float v = t[(size_t)r * C + c];
    s += v; s2 += v * v;
  }
  atomicAdd(&sums[c], s);
  atomicAdd(&sums[C + c], s2);
}
__global__ void k_bn_apply(const float* __restrict__ t, const float* __restrict__ sums,
                           const float* __restrict__ g, const float* __restrict__ b,
                           const float* __restrict__ res, float* __restrict__ out,
                           int Nr, int C) {
  int i = blockIdx.x * blockDim.x + threadIdx.x;
  if (i >= Nr * C) return;
  int c = i % C;
  float inv_n = 1.0f / (float)Nr;
  float mu = sums[c] * inv_n;
  float var = sums[C + c] * inv_n - mu * mu;
  float v = (t[i] - mu) * rsqrtf(var + 1e-5f) * g[c] + b[c];
  v = 0.5f * v * (1.0f + erff(v * 0.70710678118654752f));   // exact GELU
  out[i] = v + res[i];
}

// ======================= host orchestration ===============================
extern "C" void kernel_launch(void* const* d_in, const int* in_sizes, int n_in,
                              void* d_out, int out_size, void* d_ws, size_t ws_size,
                              hipStream_t stream) {
  const int N = NNODES, E = NEDGES, E2 = E2TOT;
  const float* x  = (const float*)d_in[0];
  const int*   ei = (const int*)d_in[1];
  const float* ea = (const float*)d_in[2];
  const float* proj_W = (const float*)d_in[3];
  const float* proj_b = (const float*)d_in[4];
  const float *Wl[3], *bl[3], *Wr[3], *br[3], *We[3], *att[3], *bias[3], *bn_g[3], *bn_b[3];
  for (int i = 0; i < 3; ++i) {
    int o = 5 + i * 9;
    Wl[i] = (const float*)d_in[o]; bl[i] = (const float*)d_in[o + 1];
    Wr[i] = (const float*)d_in[o + 2]; br[i] = (const float*)d_in[o + 3];
    We[i] = (const float*)d_in[o + 4]; att[i] = (const float*)d_in[o + 5];
    bias[i] = (const float*)d_in[o + 6]; bn_g[i] = (const float*)d_in[o + 7];
    bn_b[i] = (const float*)d_in[o + 8];
  }
  const float* skip_W = (const float*)d_in[32];   // layer 2 only (128 -> 32)
  const float* post_W = (const float*)d_in[33];   // 32 -> 128
  const float* post_b = (const float*)d_in[34];

  // workspace carve-up
  char* p = (char*)d_ws;
  auto alloc = [&](size_t bytes) { void* r = (void*)p; p += (bytes + 255) & ~(size_t)255; return r; };
  float*    ea2     = (float*)alloc((size_t)E2 * EDIM * 4);
  float*    deg     = (float*)alloc((size_t)N * 4);
  float*    hA      = (float*)alloc((size_t)N * HIDDIM * 4);
  float*    hB      = (float*)alloc((size_t)N * HIDDIM * 4);
  bf16_t*   h_bf    = (bf16_t*)alloc((size_t)N * HIDDIM * 2);
  float*    xl      = (float*)alloc((size_t)N * HIDDIM * 4);
  float*    xr      = (float*)alloc((size_t)N * HIDDIM * 4);
  float*    skipbuf = (float*)alloc((size_t)N * HDC * 4);
  float*    tbuf    = (float*)alloc((size_t)N * HIDDIM * 4);
  float*    logits  = (float*)alloc((size_t)E2 * NHEADS * 4);
  unsigned* maxkey  = (unsigned*)alloc((size_t)N * NHEADS * 4);
  float*    den     = (float*)alloc((size_t)N * NHEADS * 4);
  float*    sums    = (float*)alloc(2 * HIDDIM * 4);
  bf16_t*   projWt  = (bf16_t*)alloc(128 * 128 * 2);
  bf16_t*   WlT[3], *WrT[3];
  for (int i = 0; i < 3; ++i) {
    WlT[i] = (bf16_t*)alloc(128 * 128 * 2);
    WrT[i] = (bf16_t*)alloc(128 * 128 * 2);
  }
  bf16_t* skipWt = (bf16_t*)alloc(32 * 128 * 2);   // [32][128] = skip_W^T
  bf16_t* postWt = (bf16_t*)alloc(128 * 32 * 2);   // [128][32] = post_W^T

  const int TPB = 256;
  auto G = [](int n, int b) { return (n + b - 1) / b; };

  // weight prep: fp32 -> bf16 transposed
  k_w_to_bf_t<<<G(128 * 128, TPB), TPB, 0, stream>>>(proj_W, projWt, 128, 128);
  for (int i = 0; i < 3; ++i) {
    k_w_to_bf_t<<<G(128 * 128, TPB), TPB, 0, stream>>>(Wl[i], WlT[i], 128, 128);
    k_w_to_bf_t<<<G(128 * 128, TPB), TPB, 0, stream>>>(Wr[i], WrT[i], 128, 128);
  }
  k_w_to_bf_t<<<G(128 * 32, TPB), TPB, 0, stream>>>(skip_W, skipWt, 128, 32);
  k_w_to_bf_t<<<G(32 * 128, TPB), TPB, 0, stream>>>(post_W, postWt, 32, 128);

  // self-loop edge attrs (mean of incoming per dst)
  hipMemcpyAsync(ea2, ea, (size_t)E * EDIM * 4, hipMemcpyDeviceToDevice, stream);
  float* loop_ea = ea2 + (size_t)E * EDIM;
  k_fill_f32<<<G(N * EDIM, TPB), TPB, 0, stream>>>(loop_ea, N * EDIM, 0.0f);
  k_fill_f32<<<G(N, TPB), TPB, 0, stream>>>(deg, N, 0.0f);
  k_deg<<<G(E, TPB), TPB, 0, stream>>>(ei + E, deg, E);
  k_ea_acc<<<G(E * EDIM, TPB), TPB, 0, stream>>>(ei + E, ea, loop_ea, E);
  k_loop_div<<<G(N * EDIM, TPB), TPB, 0, stream>>>(loop_ea, deg, N * EDIM);

  // projection: h = x @ proj_W + proj_b   (Nc=128: 8 tiles = 2 strips of 4)
  k_f2bf<<<G(N * 128, TPB), TPB, 0, stream>>>(x, h_bf, N * 128);
  k_gemm_bf16<4><<<dim3(N / 16, 2), 32, 0, stream>>>(h_bf, projWt, proj_b, hA, N, 128, 128);

  float* hcur = hA;
  float* hnext = hB;
  const int wavesPerBlk = TPB / 32;
  const int edgeGrid = G(E2, wavesPerBlk);

  for (int L = 0; L < 3; ++L) {
    const int concat = (L < 2) ? 1 : 0;
    const int Cout = concat ? HIDDIM : HDC;

    k_f2bf<<<G(N * 128, TPB), TPB, 0, stream>>>(hcur, h_bf, N * 128);
    k_gemm_bf16<4><<<dim3(N / 16, 2), 32, 0, stream>>>(h_bf, WlT[L], bl[L], xl, N, 128, 128);
    k_gemm_bf16<4><<<dim3(N / 16, 2), 32, 0, stream>>>(h_bf, WrT[L], br[L], xr, N, 128, 128);
    if (L == 2)   // skip connection 128 -> 32 (2 tiles, one strip)
      k_gemm_bf16<2><<<dim3(N / 16, 1), 32, 0, stream>>>(h_bf, skipWt, nullptr, skipbuf, N, 128, 32);

    // attention + segment softmax + aggregation
    k_fill_u32<<<G(N * NHEADS, TPB), TPB, 0, stream>>>(maxkey, N * NHEADS, ORD_NEG_INF);
    k_edge_logits<<<edgeGrid, TPB, 0, stream>>>(ei, ea2, We[L], xl, xr, att[L],
                                                logits, maxkey, E, E2);
    k_fill_f32<<<G(N * NHEADS, TPB), TPB, 0, stream>>>(den, N * NHEADS, 0.0f);
    k_edge_exp<<<G(E2 * NHEADS, TPB), TPB, 0, stream>>>(ei, logits, maxkey, den, E, E2);
    k_prefill_bias<<<G(N * Cout, TPB), TPB, 0, stream>>>(tbuf, bias[L], N * Cout, Cout);
    k_edge_aggr<<<edgeGrid, TPB, 0, stream>>>(ei, logits, den, xl, tbuf, E, E2, concat);

    // BatchNorm (batch stats) + GELU + residual
    k_fill_f32<<<G(2 * Cout, TPB), TPB, 0, stream>>>(sums, 2 * Cout, 0.0f);
    k_bnstat<<<512, TPB, 0, stream>>>(tbuf, sums, N, Cout);
    const float* res = (L < 2) ? hcur : skipbuf;
    k_bn_apply<<<G(N * Cout, TPB), TPB, 0, stream>>>(tbuf, sums, bn_g[L], bn_b[L],
                                                     res, hnext, N, Cout);
    float* tmp = hcur; hcur = hnext; hnext = tmp;
  }

  // post projection: 32 -> 128 into d_out
  k_f2bf<<<G(N * 32, TPB), TPB, 0, stream>>>(hcur, h_bf, N * 32);
  k_gemm_bf16<4><<<dim3(N / 16, 2), 32, 0, stream>>>(h_bf, postWt, post_b, (float*)d_out,
                                                     N, 32, 128);
  (void)in_sizes; (void)n_in; (void)out_size; (void)ws_size;
}
